// Net_71528385347890
// MI455X (gfx1250) — compile-verified, matrix operand order checked by reading, not run on hardware
//
#include <hip/hip_runtime.h>

typedef __attribute__((ext_vector_type(16))) __bf16 v16bf;
typedef __attribute__((ext_vector_type(8)))  float  v8f;

#define B_ROWS 8192
#define HDIM   1024
#define BM 256          // workgroup tile rows
#define BN 128          // workgroup tile cols
#define BK 32           // K per LDS stage (one WMMA K)
#define NK (HDIM / BK)  // 32 K-steps
#define LDS_STRIDE 40   // 80B row pitch -> conflict-free b128 LDS reads

static __device__ __forceinline__ unsigned short f32_to_bf16(float f) {
    union { float f; unsigned u; } in; in.f = f;
    unsigned u = in.u;
    u += 0x7FFFu + ((u >> 16) & 1u);            // round-to-nearest-even
    return (unsigned short)(u >> 16);
}
static __device__ __forceinline__ float bf16_to_f32(unsigned short h) {
    union { unsigned u; float f; } o; o.u = ((unsigned)h) << 16; return o.f;
}

__global__ void cvt_f32_to_bf16(const float* __restrict__ src,
                                unsigned short* __restrict__ dst, int n) {
    int i = blockIdx.x * blockDim.x + threadIdx.x;
    int stride = gridDim.x * blockDim.x;
    for (; i < n; i += stride) dst[i] = f32_to_bf16(src[i]);
}

// C[b,n] = sigmoid( sum_k A[b,k] * W[n,k] + bias[n] )
// A: [8192 x 1024] bf16, W: [1024 x 1024] bf16 (row = out-neuron), per-WG tile 256x128.
// 8 waves: 4 along M x 2 along N, each wave owns a 64x64 sub-tile = 4x4 WMMA tiles.
__global__ __launch_bounds__(256)
void gemm_bias_sigmoid(const unsigned short* __restrict__ A,
                       const unsigned short* __restrict__ W,
                       const float* __restrict__ bias,
                       unsigned short* __restrict__ Out)
{
    // double-buffered tiles: 2*(256 + 128)*40*2B = 60 KB of the 320 KB WGP pool
    __shared__ __align__(16) unsigned short ldsA[2][BM * LDS_STRIDE];
    __shared__ __align__(16) unsigned short ldsW[2][BN * LDS_STRIDE];

    const int tid   = threadIdx.x;
    const int wave  = tid >> 5;           // 8 waves
    const int lane  = tid & 31;
    const int laneM = lane & 15;
    const int hi    = lane >> 4;

    const int blockM = blockIdx.y * BM;
    const int blockN = blockIdx.x * BN;

    const int waveM = (wave & 3) * 64;    // 4 waves along M (64 rows each)
    const int waveN = (wave >> 2) * 64;   // 2 waves along N (64 cols each)

    // global staging coords: A -> one 32-elem row per thread; W -> 2 threads per row
    const int wrow = tid >> 1;            // 0..127
    const int wseg = (tid & 1) * 16;      // 0 or 16 (elements)

    const unsigned short* gA = A + (size_t)(blockM + tid)  * HDIM;   // row tid of A tile
    const unsigned short* gW = W + (size_t)(blockN + wrow) * HDIM + wseg;

    // ISA 16-bit A/B fragment layout: lanes 0-15 hold K {0..7,16..23}, lanes 16-31 K {8..15,24..31}
    const int c0 = hi ? 8  : 0;
    const int c1 = hi ? 24 : 16;

    v8f acc[4][4];
    #pragma unroll
    for (int i = 0; i < 4; ++i)
        #pragma unroll
        for (int j = 0; j < 4; ++j)
            acc[i][j] = (v8f){0.f,0.f,0.f,0.f,0.f,0.f,0.f,0.f};

    uint4 sa[4], sw[2];
    // stage tile 0 into buffer 0
    {
        const uint4* pa = (const uint4*)gA;
        sa[0] = pa[0]; sa[1] = pa[1]; sa[2] = pa[2]; sa[3] = pa[3];
        const uint4* pw = (const uint4*)gW;
        sw[0] = pw[0]; sw[1] = pw[1];
        #pragma unroll
        for (int j = 0; j < 4; ++j)
            *(uint4*)&ldsA[0][tid * LDS_STRIDE + 8 * j] = sa[j];
        *(uint4*)&ldsW[0][wrow * LDS_STRIDE + wseg]     = sw[0];
        *(uint4*)&ldsW[0][wrow * LDS_STRIDE + wseg + 8] = sw[1];
    }
    __syncthreads();

    for (int kt = 0; kt < NK; ++kt) {
        const int cur = kt & 1;
        const int nxt = cur ^ 1;

        // issue next tile's global loads early (latency hidden behind 16 WMMAs)
        if (kt + 1 < NK) {
            const uint4* pa = (const uint4*)(gA + (kt + 1) * BK);
            sa[0] = pa[0]; sa[1] = pa[1]; sa[2] = pa[2]; sa[3] = pa[3];
            const uint4* pw = (const uint4*)(gW + (kt + 1) * BK);
            sw[0] = pw[0]; sw[1] = pw[1];
        }
        // prefetch the k+2 tile into cache (global_prefetch_b8)
        if (kt + 2 < NK) {
            __builtin_prefetch(gA + (kt + 2) * BK, 0, 0);
            __builtin_prefetch(gW + (kt + 2) * BK, 0, 0);
        }

        // fragments from current buffer + 16 WMMAs
        union FR { uint4 q[2]; v16bf v; };
        FR afr[4], wfr[4];
        #pragma unroll
        for (int tm = 0; tm < 4; ++tm) {
            int r = waveM + tm * 16 + laneM;
            afr[tm].q[0] = *(const uint4*)&ldsA[cur][r * LDS_STRIDE + c0];
            afr[tm].q[1] = *(const uint4*)&ldsA[cur][r * LDS_STRIDE + c1];
        }
        #pragma unroll
        for (int tn = 0; tn < 4; ++tn) {
            int r = waveN + tn * 16 + laneM;
            wfr[tn].q[0] = *(const uint4*)&ldsW[cur][r * LDS_STRIDE + c0];
            wfr[tn].q[1] = *(const uint4*)&ldsW[cur][r * LDS_STRIDE + c1];
        }
        #pragma unroll
        for (int tm = 0; tm < 4; ++tm)
            #pragma unroll
            for (int tn = 0; tn < 4; ++tn)
                acc[tm][tn] = __builtin_amdgcn_wmma_f32_16x16x32_bf16(
                    false, afr[tm].v, false, wfr[tn].v,
                    (short)0, acc[tm][tn], false, false);

        // commit next tile to the other buffer, then one barrier per K-step
        if (kt + 1 < NK) {
            #pragma unroll
            for (int j = 0; j < 4; ++j)
                *(uint4*)&ldsA[nxt][tid * LDS_STRIDE + 8 * j] = sa[j];
            *(uint4*)&ldsW[nxt][wrow * LDS_STRIDE + wseg]     = sw[0];
            *(uint4*)&ldsW[nxt][wrow * LDS_STRIDE + wseg + 8] = sw[1];
        }
        __syncthreads();
    }

    // epilogue: bias + sigmoid, store bf16
    #pragma unroll
    for (int tn = 0; tn < 4; ++tn) {
        int gcol = blockN + waveN + tn * 16 + laneM;
        float bv = bias[gcol];
        #pragma unroll
        for (int tm = 0; tm < 4; ++tm) {
            #pragma unroll
            for (int r = 0; r < 8; ++r) {
                int grow = blockM + waveM + tm * 16 + (hi ? r + 8 : r);
                float v = acc[tm][tn][r] + bv;
                v = 1.0f / (1.0f + __expf(-v));
                Out[(size_t)grow * HDIM + gcol] = f32_to_bf16(v);
            }
        }
    }
}

// out[b] = dot(h[b,:], Wout) + bout — one wave32 per row, 8 rows per block
__global__ __launch_bounds__(256)
void out_layer(const unsigned short* __restrict__ h,
               const float* __restrict__ Wout,
               const float* __restrict__ bout,
               float* __restrict__ out)
{
    int row  = blockIdx.x * 8 + (threadIdx.x >> 5);
    int lane = threadIdx.x & 31;
    float s = 0.f;
    for (int i = lane; i < HDIM; i += 32)
        s += bf16_to_f32(h[(size_t)row * HDIM + i]) * Wout[i];
    #pragma unroll
    for (int off = 16; off; off >>= 1) s += __shfl_down(s, off, 32);
    if (lane == 0) out[row] = s + bout[0];
}

extern "C" void kernel_launch(void* const* d_in, const int* in_sizes, int n_in,
                              void* d_out, int out_size, void* d_ws, size_t ws_size,
                              hipStream_t stream)
{
    const float* x    = (const float*)d_in[0];
    const float* W0   = (const float*)d_in[1];
    const float* b0   = (const float*)d_in[2];
    const float* Wh   = (const float*)d_in[3];
    const float* bh   = (const float*)d_in[4];
    const float* Wout = (const float*)d_in[5];
    const float* bout = (const float*)d_in[6];
    float* out = (float*)d_out;

    // workspace: 10x[1024x1024] bf16 weights (20 MB) + 2x[8192x1024] bf16 activations (32 MB)
    unsigned short* Wb   = (unsigned short*)d_ws;
    unsigned short* act0 = Wb + (size_t)10 * HDIM * HDIM;
    unsigned short* act1 = act0 + (size_t)B_ROWS * HDIM;

    cvt_f32_to_bf16<<<512, 256, 0, stream>>>(x,  act0, B_ROWS * HDIM);
    cvt_f32_to_bf16<<<512, 256, 0, stream>>>(W0, Wb, HDIM * HDIM);
    cvt_f32_to_bf16<<<512, 256, 0, stream>>>(Wh, Wb + (size_t)HDIM * HDIM, 9 * HDIM * HDIM);

    dim3 grid(HDIM / BN, B_ROWS / BM);   // (8, 32) = 256 workgroups
    for (int l = 0; l < 10; ++l) {
        const unsigned short* in = (l & 1) ? act1 : act0;
        unsigned short*       o  = (l & 1) ? act0 : act1;
        const unsigned short* Wl = Wb + (size_t)l * HDIM * HDIM;
        const float* bl = (l == 0) ? b0 : (bh + (size_t)(l - 1) * HDIM);
        gemm_bias_sigmoid<<<grid, 256, 0, stream>>>(in, Wl, bl, o);
    }
    // after layer 9 (odd), final activations are in act0
    out_layer<<<B_ROWS / 8, 256, 0, stream>>>(act0, Wout, bout, out);
}
